// GPT_16587163697311
// MI455X (gfx1250) — matrix operand order repaired and checked by reading, not compiled, not run on hardware
//
#include <hip/hip_runtime.h>
#include <math.h>

// Model constants
#define SV 50257
#define SD 1024
#define SL 4
#define SH 16
#define SS 2048
#define SHD 64
#define SF 4096   // MLP hidden

typedef __attribute__((ext_vector_type(16))) _Float16 v16h;
typedef __attribute__((ext_vector_type(8)))  _Float16 v8h;
typedef __attribute__((ext_vector_type(4)))  _Float16 v4h;
typedef __attribute__((ext_vector_type(8)))  float    v8f;
typedef __attribute__((ext_vector_type(4)))  float    f4;

#if defined(__HIP_DEVICE_COMPILE__)
#if __has_builtin(__builtin_amdgcn_tensor_load_to_lds)
#define USE_TDM 1
#warning "ATHENA: TDM path enabled (__builtin_amdgcn_tensor_load_to_lds found)"
#if __has_include(<hip/amd_detail/amd_gfx1250_TDM.h>)
#warning "ATHENA: TDM 6-arg (therock) arity selected"
#else
#warning "ATHENA: TDM 5-arg (ROCm 7.2) arity selected"
#endif
#else
#warning "ATHENA: TDM builtin NOT found - manual copy fallback in use"
#endif
#endif

#ifdef USE_TDM
typedef __attribute__((ext_vector_type(4))) unsigned int u32x4;
typedef __attribute__((ext_vector_type(8))) int          i32x8;
typedef __attribute__((ext_vector_type(4))) int          i32x4;
#endif

__device__ __forceinline__ v16h mk16(v8h lo, v8h hi) {
    v16h r;
#pragma unroll
    for (int i = 0; i < 8; ++i) { r[i] = lo[i]; r[8 + i] = hi[i]; }
    return r;
}

__device__ __forceinline__ v8f wmma_f16(v16h a, v16h b, v8f c) {
    return __builtin_amdgcn_wmma_f32_16x16x32_f16(
        false, a, false, b, (short)0, c, false, false);
}

__device__ __forceinline__ float gelu_exact(float x) {
    return 0.5f * x * (1.0f + erff(x * 0.70710678118654752f));
}

#ifdef USE_TDM
// Issue one TDM 2D tile load: rows x 32 f16 elements, row stride = strideElems,
// into LDS at lds_off with 16B padding after each 64B row (stride-40-halves layout).
__device__ __forceinline__ void tdm_load_tile(unsigned lds_off, const _Float16* gaddr,
                                              unsigned rows, unsigned strideElems) {
    unsigned long long ga = (unsigned long long)(const void*)gaddr;
    u32x4 g0;
    g0[0] = 1u;                                        // count=1, user descriptor
    g0[1] = lds_off;                                   // lds_addr (bytes)
    g0[2] = (unsigned)(ga & 0xFFFFFFFFu);              // global_addr[31:0]
    g0[3] = (unsigned)((ga >> 32) & 0x1FFFFFFu) | (2u << 30);  // addr[56:32] | type=2
    unsigned td0 = strideElems;                        // tensor_dim0
    unsigned td1 = rows;                               // tensor_dim1
    i32x8 g1;
    // data_size=1 (2B) | pad_enable | pad_interval=3 (16 DW) | pad_amount=3 (4 DW)
    g1[0] = (int)((1u << 16) | (1u << 20) | (3u << 22) | (3u << 25));
    g1[1] = (int)((td0 & 0xFFFFu) << 16);              // [63:48] = tensor_dim0 lo
    g1[2] = (int)(((td0 >> 16) & 0xFFFFu) | ((td1 & 0xFFFFu) << 16));
    g1[3] = (int)(((td1 >> 16) & 0xFFFFu) | (32u << 16));   // tile_dim0 = 32
    g1[4] = (int)(rows & 0xFFFFu);                     // tile_dim1 = rows, tile_dim2 = 0
    g1[5] = (int)strideElems;                          // tensor_dim0_stride[31:0]
    g1[6] = 0;
    g1[7] = 0;
    i32x4 gz = {};
#if __has_include(<hip/amd_detail/amd_gfx1250_TDM.h>)
    i32x8 gz8 = {};
    __builtin_amdgcn_tensor_load_to_lds(g0, g1, gz, gz, gz8, 0);
#else
    __builtin_amdgcn_tensor_load_to_lds(g0, g1, gz, gz, 0);
#endif
}
#endif

// ---------------------------------------------------------------------------
// fp32 -> f16 convert (one thread per float4)
// ---------------------------------------------------------------------------
__global__ __launch_bounds__(256) void cvt_kernel(const float* __restrict__ src,
                                                  _Float16* __restrict__ dst, long n4) {
    long i = (long)blockIdx.x * 256 + threadIdx.x;
    if (i >= n4) return;
    f4 v = ((const f4*)src)[i];
    v4h h;
#pragma unroll
    for (int j = 0; j < 4; ++j) h[j] = (_Float16)v[j];
    ((v4h*)dst)[i] = h;
}

// ---------------------------------------------------------------------------
// fp32 [K][N] -> f16 [N][K] transpose-convert (32x32 LDS tiles, z = layer)
// ---------------------------------------------------------------------------
__global__ __launch_bounds__(256) void cvtT_kernel(const float* __restrict__ src,
                                                   _Float16* __restrict__ dst,
                                                   int K, int N) {
    __shared__ _Float16 t[32][33];
    size_t loff = (size_t)blockIdx.z * K * N;
    src += loff; dst += loff;
    int k0 = blockIdx.y * 32, n0 = blockIdx.x * 32;
    int tr = threadIdx.x >> 5, tc = threadIdx.x & 31;
#pragma unroll
    for (int i = 0; i < 4; ++i) {
        int r = tr + i * 8;
        t[r][tc] = (_Float16)src[(size_t)(k0 + r) * N + n0 + tc];
    }
    __syncthreads();
#pragma unroll
    for (int i = 0; i < 4; ++i) {
        int r = tr + i * 8;
        dst[(size_t)(n0 + r) * K + k0 + tc] = t[tc][r];
    }
}

// ---------------------------------------------------------------------------
// Embedding gather (vectorized)
// ---------------------------------------------------------------------------
__global__ __launch_bounds__(256) void embed_kernel(const int* __restrict__ ids,
                                                    const float* __restrict__ emb,
                                                    float* __restrict__ x) {
    int s = blockIdx.x;
    int id = ids[s];
    const f4* src = (const f4*)(emb + (size_t)id * SD);
    f4* dst = (f4*)(x + (size_t)s * SD);
    dst[threadIdx.x] = src[threadIdx.x];
}

// ---------------------------------------------------------------------------
// LayerNorm: one block per row, f16 output (feeds GEMM A)
// ---------------------------------------------------------------------------
__global__ __launch_bounds__(256) void ln_kernel(const float* __restrict__ x,
                                                 const float* __restrict__ g,
                                                 const float* __restrict__ b,
                                                 _Float16* __restrict__ out) {
    __shared__ float red[256];
    int r = blockIdx.x, tid = threadIdx.x;
    const float* row = x + (size_t)r * SD;
    f4 xv = *(const f4*)&row[tid * 4];
    red[tid] = xv[0] + xv[1] + xv[2] + xv[3];
    __syncthreads();
    for (int st = 128; st > 0; st >>= 1) { if (tid < st) red[tid] += red[tid + st]; __syncthreads(); }
    float mean = red[0] * (1.0f / SD);
    __syncthreads();
    float vs = 0.f;
#pragma unroll
    for (int j = 0; j < 4; ++j) { float d = xv[j] - mean; vs += d * d; }
    red[tid] = vs;
    __syncthreads();
    for (int st = 128; st > 0; st >>= 1) { if (tid < st) red[tid] += red[tid + st]; __syncthreads(); }
    float inv = rsqrtf(red[0] * (1.0f / SD) + 1e-5f);
    f4 gv = *(const f4*)&g[tid * 4];
    f4 bv = *(const f4*)&b[tid * 4];
    v4h ov;
#pragma unroll
    for (int j = 0; j < 4; ++j) ov[j] = (_Float16)((xv[j] - mean) * inv * gv[j] + bv[j]);
    *(v4h*)&out[(size_t)r * SD + tid * 4] = ov;
}

// ---------------------------------------------------------------------------
// RoPE in-place on q and k (fp32)
// ---------------------------------------------------------------------------
__global__ __launch_bounds__(256) void rope_kernel(float* __restrict__ q, float* __restrict__ k) {
    int idx = blockIdx.x * blockDim.x + threadIdx.x;
    if (idx >= SS * SH * 32) return;
    int i = idx & 31;
    int h = (idx >> 5) & (SH - 1);
    int s = idx >> 9;
    float fr = powf(10000.0f, -(float)i * (1.0f / 32.0f));
    float ang = (float)s * fr;
    float c = cosf(ang), sn = sinf(ang);
    size_t base = (size_t)s * SD + h * SHD + i;
    float a0 = q[base], a1 = q[base + 32];
    q[base] = a0 * c - a1 * sn;
    q[base + 32] = a1 * c + a0 * sn;
    float b0 = k[base], b1 = k[base + 32];
    k[base] = b0 * c - b1 * sn;
    k[base + 32] = b1 * c + b0 * sn;
}

// ---------------------------------------------------------------------------
// WMMA GEMM (all-f16 operands): C = act(A[M,K] @ B^T + bias) + residual
//   A: f16 [M][K].  B: f16 [N][K] (pre-transposed weights / emb).
//   Output: f32 (C32) or f16 (C16) selected by C16 != nullptr.
//   Block tile 128x64, 8 waves, wave tile 32x32, K-step 32.
//   Full tiles: TDM double-buffered pipeline (or vector-copy fallback).
// ---------------------------------------------------------------------------
__global__ __launch_bounds__(256) void gemm_wmma(const _Float16* __restrict__ A,
                                                 const _Float16* __restrict__ B,
                                                 const float* __restrict__ bias,
                                                 const float* __restrict__ residual,
                                                 float* __restrict__ C32,
                                                 _Float16* __restrict__ C16,
                                                 int M, int N, int K, int act) {
    __shared__ _Float16 ldsA[2][128 * 40];  // [m][k] stride 40 halves (80B rows)
    __shared__ _Float16 ldsB[2][64 * 40];   // [n][k]
    const int tid = threadIdx.x;
    const int lane = tid & 31, wave = tid >> 5;
    const int lg = lane >> 4, lm = lane & 15;
    const int m0 = blockIdx.y * 128, n0 = blockIdx.x * 64;
    const int wm = (wave & 3) * 32;
    const int wn = (wave >> 2) * 32;

    v8f acc[2][2] = {};
    const bool full = (m0 + 128 <= M) && (n0 + 64 <= N);

    if (full) {
#ifdef USE_TDM
        // wave id as a true SGPR so the TDM-issue branch is scalar
        const int wave_s = __builtin_amdgcn_readfirstlane(wave);
        const unsigned ldsA_off[2] = { (unsigned)(size_t)(void*)&ldsA[0][0],
                                       (unsigned)(size_t)(void*)&ldsA[1][0] };
        const unsigned ldsB_off[2] = { (unsigned)(size_t)(void*)&ldsB[0][0],
                                       (unsigned)(size_t)(void*)&ldsB[1][0] };
        if (wave_s == 0) {
            tdm_load_tile(ldsA_off[0], A + (size_t)m0 * K, 128, (unsigned)K);
            tdm_load_tile(ldsB_off[0], B + (size_t)n0 * K, 64,  (unsigned)K);
        }
        int buf = 0;
        for (int k0 = 0; k0 < K; k0 += 32, buf ^= 1) {
            if (wave_s == 0) {
                if (k0 + 32 < K) {
                    tdm_load_tile(ldsA_off[buf ^ 1], A + (size_t)m0 * K + k0 + 32, 128, (unsigned)K);
                    tdm_load_tile(ldsB_off[buf ^ 1], B + (size_t)n0 * K + k0 + 32, 64,  (unsigned)K);
                    __builtin_amdgcn_s_wait_tensorcnt(2);
                } else {
                    __builtin_amdgcn_s_wait_tensorcnt(0);
                }
            }
            __syncthreads();   // current buffer LDS visible to all waves
            v16h af[2], bf[2];
#pragma unroll
            for (int ma = 0; ma < 2; ++ma) {
                int mr = wm + ma * 16 + lm;
                v8h alo = *(const v8h*)&ldsA[buf][mr * 40 + lg * 8];
                v8h ahi = *(const v8h*)&ldsA[buf][mr * 40 + 16 + lg * 8];
                af[ma] = mk16(alo, ahi);
            }
#pragma unroll
            for (int t = 0; t < 2; ++t) {
                int nn = wn + t * 16 + lm;
                v8h blo = *(const v8h*)&ldsB[buf][nn * 40 + lg * 8];
                v8h bhi = *(const v8h*)&ldsB[buf][nn * 40 + 16 + lg * 8];
                bf[t] = mk16(blo, bhi);
            }
#pragma unroll
            for (int ma = 0; ma < 2; ++ma)
#pragma unroll
                for (int t = 0; t < 2; ++t)
                    acc[ma][t] = wmma_f16(af[ma], bf[t], acc[ma][t]);
            __syncthreads();   // reads done before this buffer is overwritten
        }
#else
        // Manual pipelined copy fallback: A 16 halves/thread, B 8 halves/thread
        const int ar = tid >> 1, ac = (tid & 1) * 16;
        const int br = tid >> 2, bc = (tid & 3) * 8;
        v8h pa0, pa1, pb;
        auto loadT = [&](int kk) {
            pa0 = *(const v8h*)&A[(size_t)(m0 + ar) * K + kk + ac];
            pa1 = *(const v8h*)&A[(size_t)(m0 + ar) * K + kk + ac + 8];
            pb  = *(const v8h*)&B[(size_t)(n0 + br) * K + kk + bc];
        };
        loadT(0);
        for (int k0 = 0; k0 < K; k0 += 32) {
            *(v8h*)&ldsA[0][ar * 40 + ac]     = pa0;
            *(v8h*)&ldsA[0][ar * 40 + ac + 8] = pa1;
            *(v8h*)&ldsB[0][br * 40 + bc]     = pb;
            __syncthreads();
            if (k0 + 32 < K) loadT(k0 + 32);
            v16h af[2], bf[2];
#pragma unroll
            for (int ma = 0; ma < 2; ++ma) {
                int mr = wm + ma * 16 + lm;
                v8h alo = *(const v8h*)&ldsA[0][mr * 40 + lg * 8];
                v8h ahi = *(const v8h*)&ldsA[0][mr * 40 + 16 + lg * 8];
                af[ma] = mk16(alo, ahi);
            }
#pragma unroll
            for (int t = 0; t < 2; ++t) {
                int nn = wn + t * 16 + lm;
                v8h blo = *(const v8h*)&ldsB[0][nn * 40 + lg * 8];
                v8h bhi = *(const v8h*)&ldsB[0][nn * 40 + 16 + lg * 8];
                bf[t] = mk16(blo, bhi);
            }
#pragma unroll
            for (int ma = 0; ma < 2; ++ma)
#pragma unroll
                for (int t = 0; t < 2; ++t)
                    acc[ma][t] = wmma_f16(af[ma], bf[t], acc[ma][t]);
            __syncthreads();
        }
#endif
    } else {
        // guarded edge path (only the logits GEMM's last N-block)
        for (int k0 = 0; k0 < K; k0 += 32) {
            __syncthreads();
            for (int idx = tid; idx < 128 * 32; idx += 256) {
                int r = idx >> 5, c = idx & 31;
                int gm = m0 + r;
                ldsA[0][r * 40 + c] = (gm < M) ? A[(size_t)gm * K + k0 + c] : (_Float16)0.f;
            }
            for (int idx = tid; idx < 64 * 32; idx += 256) {
                int n = idx >> 5, c = idx & 31;
                int gn = n0 + n;
                ldsB[0][n * 40 + c] = (gn < N) ? B[(size_t)gn * K + k0 + c] : (_Float16)0.f;
            }
            __syncthreads();
            v16h af[2], bf[2];
#pragma unroll
            for (int ma = 0; ma < 2; ++ma) {
                int mr = wm + ma * 16 + lm;
                v8h alo = *(const v8h*)&ldsA[0][mr * 40 + lg * 8];
                v8h ahi = *(const v8h*)&ldsA[0][mr * 40 + 16 + lg * 8];
                af[ma] = mk16(alo, ahi);
            }
#pragma unroll
            for (int t = 0; t < 2; ++t) {
                int nn = wn + t * 16 + lm;
                v8h blo = *(const v8h*)&ldsB[0][nn * 40 + lg * 8];
                v8h bhi = *(const v8h*)&ldsB[0][nn * 40 + 16 + lg * 8];
                bf[t] = mk16(blo, bhi);
            }
#pragma unroll
            for (int ma = 0; ma < 2; ++ma)
#pragma unroll
                for (int t = 0; t < 2; ++t)
                    acc[ma][t] = wmma_f16(af[ma], bf[t], acc[ma][t]);
        }
    }

    // epilogue: C layout (m,n): lane = n + 16*(m>>3), vgpr = m&7
#pragma unroll
    for (int ma = 0; ma < 2; ++ma)
#pragma unroll
        for (int t = 0; t < 2; ++t)
#pragma unroll
            for (int vv = 0; vv < 8; ++vv) {
                int gm = m0 + wm + ma * 16 + vv + 8 * lg;
                int gn = n0 + wn + t * 16 + lm;
                if (gm < M && gn < N) {
                    float val = acc[ma][t][vv];
                    if (bias) val += bias[gn];
                    if (act) val = gelu_exact(val);
                    if (residual) val += residual[(size_t)gm * N + gn];
                    if (C16) C16[(size_t)gm * N + gn] = (_Float16)val;
                    else     C32[(size_t)gm * N + gn] = val;
                }
            }
}

// ---------------------------------------------------------------------------
// Flash attention: grid (S/16, H), 1 wave per block; f16 output (feeds GEMM A)
// ---------------------------------------------------------------------------
__global__ __launch_bounds__(32) void attn_kernel(const float* __restrict__ q,
                                                  const float* __restrict__ k,
                                                  const float* __restrict__ v,
                                                  _Float16* __restrict__ o) {
    __shared__ _Float16 ldsK[32 * 72];
    __shared__ _Float16 ldsVT[64 * 40];
    __shared__ _Float16 ldsP[16 * 40];
    __shared__ float    ldsS[16 * 36];
    __shared__ float rowmax[16], rowsum[16], rowfac[16];

    const int qt = blockIdx.x, hh = blockIdx.y;
    const int lane = threadIdx.x;
    const int lg = lane >> 4, lm = lane & 15;
    const int qbase = qt * 16;
    const size_t hoff = (size_t)hh * SHD;
    const float scale = 0.125f;

    v16h aq[2];
#pragma unroll
    for (int ks = 0; ks < 2; ++ks) {
        const float* qrow = q + (size_t)(qbase + lm) * SD + hoff + ks * 32;
        f4 q0 = *(const f4*)&qrow[lg * 8];
        f4 q1 = *(const f4*)&qrow[lg * 8 + 4];
        f4 q2 = *(const f4*)&qrow[16 + lg * 8];
        f4 q3 = *(const f4*)&qrow[16 + lg * 8 + 4];
        v16h a;
#pragma unroll
        for (int s = 0; s < 4; ++s) {
            a[s]      = (_Float16)(q0[s] * scale);
            a[4 + s]  = (_Float16)(q1[s] * scale);
            a[8 + s]  = (_Float16)(q2[s] * scale);
            a[12 + s] = (_Float16)(q3[s] * scale);
        }
        aq[ks] = a;
    }

    if (lane < 16) { rowmax[lane] = -3e38f; rowsum[lane] = 0.f; }
    v8f oacc[4] = {};

    const int nkb = (qbase + 16 + 31) >> 5;
    for (int kb = 0; kb < nkb; ++kb) {
        const int kbase = kb * 32;
        __syncthreads();
        for (int idx = lane; idx < 512; idx += 32) {
            int r = idx >> 4, c4 = (idx & 15) * 4;
            size_t base = (size_t)(kbase + r) * SD + hoff + c4;
            f4 kv = *(const f4*)&k[base];
            f4 vv = *(const f4*)&v[base];
            v4h kh;
#pragma unroll
            for (int j = 0; j < 4; ++j) kh[j] = (_Float16)kv[j];
            *(v4h*)&ldsK[r * 72 + c4] = kh;
#pragma unroll
            for (int j = 0; j < 4; ++j) ldsVT[(c4 + j) * 40 + r] = (_Float16)vv[j];
        }
        __syncthreads();

        v8f sacc[2] = {};
#pragma unroll
        for (int nt = 0; nt < 2; ++nt) {
#pragma unroll
            for (int ks = 0; ks < 2; ++ks) {
                v8h blo = *(const v8h*)&ldsK[(nt * 16 + lm) * 72 + ks * 32 + lg * 8];
                v8h bhi = *(const v8h*)&ldsK[(nt * 16 + lm) * 72 + ks * 32 + 16 + lg * 8];
                sacc[nt] = wmma_f16(aq[ks], mk16(blo, bhi), sacc[nt]);
            }
        }

#pragma unroll
        for (int nt = 0; nt < 2; ++nt) {
#pragma unroll
            for (int vv = 0; vv < 8; ++vv) {
                int m = vv + 8 * lg;
                int query = qbase + m;
                int key = kbase + nt * 16 + lm;
                float sv = (key <= query) ? sacc[nt][vv] : -3e38f;
                ldsS[m * 36 + nt * 16 + lm] = sv;
            }
        }
        __syncthreads();

        if (lane < 16) {
            int m = lane;
            float om = rowmax[m], nm = om;
            for (int j = 0; j < 32; ++j) nm = fmaxf(nm, ldsS[m * 36 + j]);
            float fac, psum = 0.f;
            if (nm <= -1e37f) {
                fac = 1.f;
                for (int j = 0; j < 32; ++j) ldsP[m * 40 + j] = (_Float16)0.f;
            } else {
                fac = expf(om - nm);
                for (int j = 0; j < 32; ++j) {
                    float p = expf(ldsS[m * 36 + j] - nm);
                    psum += p;
                    ldsP[m * 40 + j] = (_Float16)p;
                }
            }
            rowsum[m] = rowsum[m] * fac + psum;
            rowmax[m] = nm;
            rowfac[m] = fac;
        }
        __syncthreads();

        float facv[8];
#pragma unroll
        for (int vv = 0; vv < 8; ++vv) facv[vv] = rowfac[vv + 8 * lg];
#pragma unroll
        for (int t = 0; t < 4; ++t)
#pragma unroll
            for (int vv = 0; vv < 8; ++vv) oacc[t][vv] *= facv[vv];

        v8h plo = *(const v8h*)&ldsP[lm * 40 + lg * 8];
        v8h phi = *(const v8h*)&ldsP[lm * 40 + 16 + lg * 8];
        v16h ap = mk16(plo, phi);

#pragma unroll
        for (int t = 0; t < 4; ++t) {
            v8h blo = *(const v8h*)&ldsVT[(t * 16 + lm) * 40 + lg * 8];
            v8h bhi = *(const v8h*)&ldsVT[(t * 16 + lm) * 40 + 16 + lg * 8];
            oacc[t] = wmma_f16(ap, mk16(blo, bhi), oacc[t]);
        }
    }

    __syncthreads();
    float sums[8];
#pragma unroll
    for (int vv = 0; vv < 8; ++vv) sums[vv] = rowsum[vv + 8 * lg];
#pragma unroll
    for (int t = 0; t < 4; ++t)
#pragma unroll
        for (int vv = 0; vv < 8; ++vv) {
            int m = vv + 8 * lg;
            int n = t * 16 + lm;
            o[(size_t)(qbase + m) * SD + hoff + n] = (_Float16)(oacc[t][vv] / sums[vv]);
        }
}

// ---------------------------------------------------------------------------
// Host-side orchestration
// ---------------------------------------------------------------------------
extern "C" void kernel_launch(void* const* d_in, const int* in_sizes, int n_in,
                              void* d_out, int out_size, void* d_ws, size_t ws_size,
                              hipStream_t stream) {
    const int*   ids   = (const int*)d_in[0];
    const float* emb   = (const float*)d_in[1];
    const float* Wq    = (const float*)d_in[2];
    const float* Wk    = (const float*)d_in[3];
    const float* Wv    = (const float*)d_in[4];
    const float* Wo    = (const float*)d_in[5];
    const float* ln1_g = (const float*)d_in[6];
    const float* ln1_b = (const float*)d_in[7];
    const float* ln2_g = (const float*)d_in[8];
    const float* ln2_b = (const float*)d_in[9];
    const float* W1    = (const float*)d_in[10];
    const float* b1    = (const float*)d_in[11];
    const float* W2    = (const float*)d_in[12];
    const float* b2    = (const float*)d_in[13];
    const float* lnf_g = (const float*)d_in[14];
    const float* lnf_b = (const float*)d_in[15];
    float* out = (float*)d_out;

    // fp32 region
    float* x    = (float*)d_ws;                          // S*D
    float* q    = x    + (size_t)SS * SD;
    float* kbuf = q    + (size_t)SS * SD;
    float* vbuf = kbuf + (size_t)SS * SD;
    // f16 region
    _Float16* h16   = (_Float16*)(vbuf + (size_t)SS * SD);
    _Float16* att16 = h16   + (size_t)SS * SD;
    _Float16* mid16 = att16 + (size_t)SS * SD;           // S*F
    _Float16* emb16 = mid16 + (size_t)SS * SF;           // V*D
    _Float16* wqT   = emb16 + (size_t)SV * SD;           // L*D*D each
    _Float16* wkT   = wqT   + (size_t)SL * SD * SD;
    _Float16* wvT   = wkT   + (size_t)SL * SD * SD;
    _Float16* woT   = wvT   + (size_t)SL * SD * SD;
    _Float16* w1T   = woT   + (size_t)SL * SD * SD;      // L*D*F
    _Float16* w2T   = w1T   + (size_t)SL * SD * SF;      // L*F*D

    dim3 blk256(256), blk32(32);
    dim3 gD((SD + 63) / 64, (SS + 127) / 128);
    dim3 gF((SF + 63) / 64, (SS + 127) / 128);
    dim3 gV((SV + 63) / 64, (SS + 127) / 128);
    dim3 gAttn(SS / 16, SH);
    int ropeN = SS * SH * 32;

    // ---- one-time (per call) weight conversion/transposition ----
    long embN4 = ((long)SV * SD) / 4;
    cvt_kernel<<<(unsigned)((embN4 + 255) / 256), blk256, 0, stream>>>(emb, emb16, embN4);
    dim3 gTdd(SD / 32, SD / 32, SL);
    cvtT_kernel<<<gTdd, blk256, 0, stream>>>(Wq, wqT, SD, SD);
    cvtT_kernel<<<gTdd, blk256, 0, stream>>>(Wk, wkT, SD, SD);
    cvtT_kernel<<<gTdd, blk256, 0, stream>>>(Wv, wvT, SD, SD);
    cvtT_kernel<<<gTdd, blk256, 0, stream>>>(Wo, woT, SD, SD);
    dim3 gT1(SF / 32, SD / 32, SL);
    cvtT_kernel<<<gT1, blk256, 0, stream>>>(W1, w1T, SD, SF);
    dim3 gT2(SD / 32, SF / 32, SL);
    cvtT_kernel<<<gT2, blk256, 0, stream>>>(W2, w2T, SF, SD);

    embed_kernel<<<SS, blk256, 0, stream>>>(ids, emb, x);

    for (int l = 0; l < SL; ++l) {
        const _Float16* wq = wqT + (size_t)l * SD * SD;
        const _Float16* wk = wkT + (size_t)l * SD * SD;
        const _Float16* wv = wvT + (size_t)l * SD * SD;
        const _Float16* wo = woT + (size_t)l * SD * SD;
        const _Float16* w1 = w1T + (size_t)l * SD * SF;
        const _Float16* w2 = w2T + (size_t)l * SF * SD;

        ln_kernel<<<SS, blk256, 0, stream>>>(x, ln1_g + l * SD, ln1_b + l * SD, h16);
        gemm_wmma<<<gD, blk256, 0, stream>>>(h16, wq, nullptr, nullptr, q,    nullptr, SS, SD, SD, 0);
        gemm_wmma<<<gD, blk256, 0, stream>>>(h16, wk, nullptr, nullptr, kbuf, nullptr, SS, SD, SD, 0);
        gemm_wmma<<<gD, blk256, 0, stream>>>(h16, wv, nullptr, nullptr, vbuf, nullptr, SS, SD, SD, 0);
        rope_kernel<<<(ropeN + 255) / 256, blk256, 0, stream>>>(q, kbuf);
        attn_kernel<<<gAttn, blk32, 0, stream>>>(q, kbuf, vbuf, att16);
        gemm_wmma<<<gD, blk256, 0, stream>>>(att16, wo, nullptr, x, x, nullptr, SS, SD, SD, 0);

        ln_kernel<<<SS, blk256, 0, stream>>>(x, ln2_g + l * SD, ln2_b + l * SD, h16);
        gemm_wmma<<<gF, blk256, 0, stream>>>(h16, w1, b1 + l * SF, nullptr, nullptr, mid16, SS, SF, SD, 1);
        gemm_wmma<<<gD, blk256, 0, stream>>>(mid16, w2, b2 + l * SD, x, x, nullptr, SS, SD, SF, 0);
    }

    ln_kernel<<<SS, blk256, 0, stream>>>(x, lnf_g, lnf_b, h16);
    // logits = h @ emb^T : emb16 is [V][D] row-major == B^T layout
    gemm_wmma<<<gV, blk256, 0, stream>>>(h16, emb16, nullptr, nullptr, out, nullptr, SS, SV, SD, 0);
}
